// EnhancedEURLTransformer_87694642249910
// MI455X (gfx1250) — compile-verified, matrix operand order checked by reading, not run on hardware
//
#include <hip/hip_runtime.h>
#include <hip/hip_bf16.h>

// ---------------- problem constants ----------------
#define BB 2
#define SS 2048
#define DD 512
#define HH 8
#define RR 64
#define LL 6
#define VV 32000
#define FFF 2048
#define HD 64
#define BS (BB*SS)
#define KTOP 409
#define LN_EPS 1e-5f

// feature gates (compile-only loop: flip if a path fails to assemble)
#define TRY_ASYNC_ASM 1
#if defined(__has_builtin)
# if __has_builtin(__builtin_amdgcn_tensor_load_to_lds)
#  define HAVE_TDM 1
# endif
#endif
#ifndef HAVE_TDM
# define HAVE_TDM 0
#endif

typedef _Float16 f16;
typedef unsigned int u32;
typedef __attribute__((ext_vector_type(16))) _Float16 v16h;
typedef __attribute__((ext_vector_type(8)))  _Float16 v8h;
typedef __attribute__((ext_vector_type(8)))  float    v8f;
typedef __attribute__((ext_vector_type(4)))  float    v4f;
typedef __attribute__((ext_vector_type(4)))  u32      u32x4;
typedef __attribute__((ext_vector_type(4)))  int      i32x4;
typedef __attribute__((ext_vector_type(8)))  int      i32x8;

// -------- WMMA fragment loaders (wave32 A/B layout: row = lane%16,
// halves 0..7 -> k = kb + (lane/16)*8, halves 8..15 -> k = kb+16+(lane/16)*8) --------
__device__ inline v16h load_frag_f16(const f16* p) {
  v8h lo = *(const v8h*)p;
  v8h hi = *(const v8h*)(p + 16);
  v16h r;
#pragma unroll
  for (int i = 0; i < 8; ++i) { r[i] = lo[i]; r[i + 8] = hi[i]; }
  return r;
}

__device__ inline v16h load_frag_f32(const float* p) {
  v4f a0 = *(const v4f*)p;
  v4f a1 = *(const v4f*)(p + 4);
  v4f b0 = *(const v4f*)(p + 16);
  v4f b1 = *(const v4f*)(p + 20);
  v16h r;
#pragma unroll
  for (int i = 0; i < 4; ++i) {
    r[i]      = (f16)a0[i];
    r[i + 4]  = (f16)a1[i];
    r[i + 8]  = (f16)b0[i];
    r[i + 12] = (f16)b1[i];
  }
  return r;
}

// -------- TDM: one 2D-tile descriptor load (rows x rowlen f16, row stride in elems) --------
__device__ inline void tdm_load_tile_f16(const f16* gsrc, u32 lds_off,
                                         int rows, int rowlen, int stride) {
#if HAVE_TDM
  unsigned long long ga = (unsigned long long)(size_t)gsrc;
  u32 dim0 = (u32)stride;
  u32x4 g0;
  g0[0] = 1u;                                   // count=1, user descriptor
  g0[1] = lds_off;                              // lds_addr (bytes)
  g0[2] = (u32)(ga & 0xffffffffu);              // global_addr[31:0]
  g0[3] = (u32)((ga >> 32) & 0x01ffffffu) | 0x80000000u; // addr[56:32] | type=2
  i32x8 g1;
  g1[0] = 0x00010000;                           // data_size=1 (2 bytes), no multicast
  g1[1] = (int)((dim0 & 0xffffu) << 16);        // tensor_dim0 low16
  g1[2] = (int)(((dim0 >> 16) & 0xffffu) | (((u32)rows & 0xffffu) << 16)); // dim0 hi | tensor_dim1 lo
  g1[3] = (int)(((u32)rowlen & 0xffffu) << 16); // tensor_dim1 hi=0 | tile_dim0
  g1[4] = (int)((u32)rows & 0xffffu);           // tile_dim1 | tile_dim2=0
  g1[5] = (int)stride;                          // tensor_dim0_stride low32
  g1[6] = 0;                                    // stride hi | dim1_stride lo
  g1[7] = 0;
  i32x4 z4; z4[0] = z4[1] = z4[2] = z4[3] = 0;
# if __clang_major__ >= 23
  i32x8 z8;
#  pragma unroll
  for (int i = 0; i < 8; ++i) z8[i] = 0;
  __builtin_amdgcn_tensor_load_to_lds(g0, g1, z4, z4, z8, 0);
# else
  __builtin_amdgcn_tensor_load_to_lds(g0, g1, z4, z4, 0);
# endif
#else
  (void)gsrc; (void)lds_off; (void)rows; (void)rowlen; (void)stride;
#endif
}

// -------- async global -> LDS 16B copy (per-lane) --------
__device__ inline void async_copy_b128(u32 lds_off, const void* gp) {
#if TRY_ASYNC_ASM
  unsigned long long ga = (unsigned long long)(size_t)gp;
  asm volatile("global_load_async_to_lds_b128 %0, %1, off"
               :: "v"(lds_off), "v"(ga) : "memory");
#else
  *(v8h*)((char*)0 + lds_off) = *(const v8h*)gp; // unreachable config
#endif
}
__device__ inline void async_wait0() {
#if TRY_ASYNC_ASM
  asm volatile("s_wait_asynccnt 0x0" ::: "memory");
#endif
}

// ---------------- generic GEMM: out[M,N] = act(scale * A_f16[M,K] @ W_f32[N,K]^T + bias) ------
// block = 256 (8 waves). Block tile = 32(M) x 512(N); wave w owns 32x64 at n0 = bIdx.y*512+w*64.
// The shared 32x64 f16 A-slab is staged in LDS (TDM descriptor load if available).
template <bool F16OUT, bool RELU>
__global__ __launch_bounds__(256) void gemm_kernel(const f16* __restrict__ A,
                                                   const float* __restrict__ W,
                                                   const float* __restrict__ bias,
                                                   void* __restrict__ out,
                                                   int M, int N, int K, float scale) {
  const int wave = threadIdx.x >> 5;
  const int lane = threadIdx.x & 31;
  const int kh   = lane >> 4;
  const int lm   = lane & 15;
  const int m0 = blockIdx.x * 32;
  const int n0 = blockIdx.y * 512 + wave * 64;
  const bool active = (n0 < N);

  __shared__ __align__(16) f16 abuf[32 * 64];

  v8f acc[2][4];
#pragma unroll
  for (int ms = 0; ms < 2; ++ms)
#pragma unroll
    for (int t = 0; t < 4; ++t)
#pragma unroll
      for (int i = 0; i < 8; ++i) acc[ms][t][i] = 0.0f;

  for (int kslab = 0; kslab < K; kslab += 64) {
    // ---- stage A slab [32 rows x 64 k] into LDS ----
#if HAVE_TDM
    if (wave == 0) {
      tdm_load_tile_f16(A + (size_t)m0 * K + kslab,
                        (u32)(size_t)&abuf[0], 32, 64, K);
      __builtin_amdgcn_s_wait_tensorcnt(0);
    }
#else
    {
      const int t = threadIdx.x;
      const int row = t >> 3, ch = t & 7;
      const f16* gp = A + (size_t)(m0 + row) * K + kslab + ch * 8;
      *(v8h*)&abuf[row * 64 + ch * 8] = *(const v8h*)gp;
    }
#endif
    __syncthreads();

    if (active) __builtin_prefetch(W + (size_t)(n0 + lm) * K + kslab + 64, 0, 1);

#pragma unroll
    for (int kb2 = 0; kb2 < 2; ++kb2) {
      v16h a0 = load_frag_f16(&abuf[lm * 64 + kb2 * 32 + kh * 8]);
      v16h a1 = load_frag_f16(&abuf[(16 + lm) * 64 + kb2 * 32 + kh * 8]);
      if (active) {
        const int kb = kslab + kb2 * 32;
#pragma unroll
        for (int t = 0; t < 4; ++t) {
          const float* Wrow = W + (size_t)(n0 + t * 16 + lm) * K + kb + kh * 8;
          v16h b = load_frag_f32(Wrow);
          acc[0][t] = __builtin_amdgcn_wmma_f32_16x16x32_f16(false, a0, false, b,
                                                             (short)0, acc[0][t], false, false);
          acc[1][t] = __builtin_amdgcn_wmma_f32_16x16x32_f16(false, a1, false, b,
                                                             (short)0, acc[1][t], false, false);
        }
      }
    }
    __syncthreads();
  }

  if (!active) return;
#pragma unroll
  for (int ms = 0; ms < 2; ++ms)
#pragma unroll
    for (int t = 0; t < 4; ++t)
#pragma unroll
      for (int r = 0; r < 8; ++r) {
        int m = m0 + ms * 16 + r + 8 * kh;
        int n = n0 + t * 16 + lm;
        float v = acc[ms][t][r] * scale;
        if (bias) v += bias[n];
        if (RELU) v = fmaxf(v, 0.0f);
        if (F16OUT) ((f16*)out)[(size_t)m * N + n] = (f16)v;
        else        ((float*)out)[(size_t)m * N + n] = v;
      }
}

// ---------------- flash attention: one wave handles a 16-query tile of one (b,h) ----------------
// qkv layout [B,S,3D] f16 (q at 0, k at D, v at 2D within last dim)
__global__ __launch_bounds__(32) void attn_kernel(const f16* __restrict__ qkv,
                                                  f16* __restrict__ aout) {
  const int lane = threadIdx.x;
  const int kh = lane >> 4;
  const int lm = lane & 15;
  const int m0 = blockIdx.x * 16;
  const int h  = blockIdx.y;
  const int b  = blockIdx.z;
  const int LD = 3 * DD;

  __shared__ __align__(16) f16 pbuf[16 * 32];
  __shared__ __align__(16) f16 vbuf[32 * 64];

  const f16* qrow = qkv + (size_t)(b * SS + m0 + lm) * LD + h * HD + kh * 8;
  v16h aq0 = load_frag_f16(qrow);
  v16h aq1 = load_frag_f16(qrow + 32);

  float mrow[8], lrow[8];
  v8f o[4];
#pragma unroll
  for (int r = 0; r < 8; ++r) { mrow[r] = -1e30f; lrow[r] = 0.0f; }
#pragma unroll
  for (int t = 0; t < 4; ++t)
#pragma unroll
    for (int i = 0; i < 8; ++i) o[t][i] = 0.0f;

  for (int kk0 = 0; kk0 < SS; kk0 += 32) {
    // kick off async V-tile staging early (32 rows x 64 f16 = 128B/lane)
    const f16* vrow = qkv + (size_t)(b * SS + kk0 + lane) * LD + 2 * DD + h * HD;
#if TRY_ASYNC_ASM
#pragma unroll
    for (int c = 0; c < 8; ++c)
      async_copy_b128((u32)(size_t)&vbuf[lane * 64 + c * 8], vrow + c * 8);
#else
#pragma unroll
    for (int c = 0; c < 8; ++c)
      *(v8h*)&vbuf[lane * 64 + c * 8] = *(const v8h*)(vrow + c * 8);
#endif

    // ---- scores: 16x32 tile = two 16x16 WMMAs, K-dim = HD = 64 (two k-steps) ----
    v8f s[2];
#pragma unroll
    for (int sub = 0; sub < 2; ++sub) {
      const f16* krow = qkv + (size_t)(b * SS + kk0 + sub * 16 + lm) * LD + DD + h * HD + kh * 8;
      v16h bk0 = load_frag_f16(krow);
      v16h bk1 = load_frag_f16(krow + 32);
      v8f c;
#pragma unroll
      for (int i = 0; i < 8; ++i) c[i] = 0.0f;
      c = __builtin_amdgcn_wmma_f32_16x16x32_f16(false, aq0, false, bk0, (short)0, c, false, false);
      c = __builtin_amdgcn_wmma_f32_16x16x32_f16(false, aq1, false, bk1, (short)0, c, false, false);
      s[sub] = c;
    }
    float alph[8];
#pragma unroll
    for (int r = 0; r < 8; ++r) {
      float s0 = s[0][r] * 0.125f;  // 1/sqrt(HD)
      float s1 = s[1][r] * 0.125f;
      float v = fmaxf(s0, s1);
#pragma unroll
      for (int off = 8; off > 0; off >>= 1) v = fmaxf(v, __shfl_xor(v, off, 16));
      float nm = fmaxf(mrow[r], v);
      alph[r] = __expf(mrow[r] - nm);
      float p0 = __expf(s0 - nm);
      float p1 = __expf(s1 - nm);
      float rs = p0 + p1;
#pragma unroll
      for (int off = 8; off > 0; off >>= 1) rs += __shfl_xor(rs, off, 16);
      lrow[r] = lrow[r] * alph[r] + rs;
      mrow[r] = nm;
      s[0][r] = p0; s[1][r] = p1;
    }
#pragma unroll
    for (int t = 0; t < 4; ++t)
#pragma unroll
      for (int r = 0; r < 8; ++r) o[t][r] *= alph[r];

    // ---- stage P (16x32) in LDS; ensure V staging landed ----
#pragma unroll
    for (int r = 0; r < 8; ++r) {
      int m = r + 8 * kh;
      pbuf[m * 32 + lm]      = (f16)s[0][r];
      pbuf[m * 32 + 16 + lm] = (f16)s[1][r];
    }
    async_wait0();
    __syncthreads();

    // ---- O += P @ V ----
    v16h ap = load_frag_f16(&pbuf[lm * 32 + kh * 8]);
#pragma unroll
    for (int t = 0; t < 4; ++t) {
      v16h bv;
#pragma unroll
      for (int j = 0; j < 16; ++j) {
        int kk = ((j < 8) ? j : j + 8) + kh * 8;
        bv[j] = vbuf[kk * 64 + t * 16 + lm];
      }
      o[t] = __builtin_amdgcn_wmma_f32_16x16x32_f16(false, ap, false, bv, (short)0, o[t], false, false);
    }
    __syncthreads();
  }

#pragma unroll
  for (int t = 0; t < 4; ++t)
#pragma unroll
    for (int r = 0; r < 8; ++r) {
      int m = m0 + r + 8 * kh;
      int n = h * HD + t * 16 + lm;
      aout[(size_t)(b * SS + m) * DD + n] = (f16)(o[t][r] / lrow[r]);
    }
}

// ---------------- top-k(409) + renormalize: one block per row of qk[BS, S] ----------------
__global__ __launch_bounds__(256) void topk_kernel(const float* __restrict__ qk,
                                                   f16* __restrict__ sp) {
  const int tid = threadIdx.x;
  const float* x = qk + (size_t)blockIdx.x * SS;
  f16* o = sp + (size_t)blockIdx.x * SS;
  __shared__ float redf[256];
  __shared__ int   redi[256];
  __shared__ float sMax, sMin, sSum;

  float mx = -3e38f, mn = 3e38f;
  for (int i = tid; i < SS; i += 256) { float v = x[i]; mx = fmaxf(mx, v); mn = fminf(mn, v); }
  redf[tid] = mx; __syncthreads();
  for (int st = 128; st > 0; st >>= 1) { if (tid < st) redf[tid] = fmaxf(redf[tid], redf[tid + st]); __syncthreads(); }
  if (tid == 0) sMax = redf[0];
  __syncthreads();
  redf[tid] = mn; __syncthreads();
  for (int st = 128; st > 0; st >>= 1) { if (tid < st) redf[tid] = fminf(redf[tid], redf[tid + st]); __syncthreads(); }
  if (tid == 0) sMin = redf[0];
  __syncthreads();

  float lo = sMin - 1e-6f, hi = sMax;
  for (int it = 0; it < 32; ++it) {
    float mid = 0.5f * (lo + hi);
    int c = 0;
    for (int i = tid; i < SS; i += 256) c += (x[i] >= mid) ? 1 : 0;
    redi[tid] = c; __syncthreads();
    for (int st = 128; st > 0; st >>= 1) { if (tid < st) redi[tid] += redi[tid + st]; __syncthreads(); }
    int tot = redi[0];
    __syncthreads();
    if (tot >= KTOP) lo = mid; else hi = mid;
  }
  const float thr = lo;
  const float M = sMax;
  float se = 0.0f;
  for (int i = tid; i < SS; i += 256) { float v = x[i]; if (v >= thr) se += __expf(v - M); }
  redf[tid] = se; __syncthreads();
  for (int st = 128; st > 0; st >>= 1) { if (tid < st) redf[tid] += redf[tid + st]; __syncthreads(); }
  if (tid == 0) sSum = redf[0];
  __syncthreads();
  const float inv = 1.0f / (sSum + 1e-9f);
  for (int i = tid; i < SS; i += 256) {
    float v = x[i];
    o[i] = (v >= thr) ? (f16)(__expf(v - M) * inv) : (f16)0.0f;
  }
}

// ---------------- fused residual + gate + LayerNorm; one block per row ----------------
__global__ __launch_bounds__(256) void lnfuse_kernel(float* __restrict__ x32,
                                                     f16* __restrict__ x16,
                                                     const float* __restrict__ add1,
                                                     const float* __restrict__ add2,
                                                     const float* __restrict__ lam,
                                                     int layer, int useGate,
                                                     const float* __restrict__ sw,
                                                     const float* __restrict__ bw) {
  const int tid = threadIdx.x;
  const size_t row = blockIdx.x;
  __shared__ float red[256];
  __shared__ float sMean, sRstd;

  float cA = 1.0f, cB = 0.0f;
  if (useGate) {
    float g = 1.0f / (1.0f + __expf(-lam[layer]));
    cA = g; cB = 1.0f - g;
  }
  const int d0 = tid, d1 = tid + 256;
  float y0 = x32[row * DD + d0] + cA * add1[row * DD + d0];
  float y1 = x32[row * DD + d1] + cA * add1[row * DD + d1];
  if (add2) {
    y0 += cB * add2[row * DD + d0];
    y1 += cB * add2[row * DD + d1];
  }
  red[tid] = y0 + y1; __syncthreads();
  for (int st = 128; st > 0; st >>= 1) { if (tid < st) red[tid] += red[tid + st]; __syncthreads(); }
  if (tid == 0) sMean = red[0] * (1.0f / DD);
  __syncthreads();
  const float mu = sMean;
  float c0 = y0 - mu, c1 = y1 - mu;
  red[tid] = c0 * c0 + c1 * c1; __syncthreads();
  for (int st = 128; st > 0; st >>= 1) { if (tid < st) red[tid] += red[tid + st]; __syncthreads(); }
  if (tid == 0) sRstd = rsqrtf(red[0] * (1.0f / DD) + LN_EPS);
  __syncthreads();
  const float rs = sRstd;
  float o0 = c0 * rs * sw[d0] + bw[d0];
  float o1 = c1 * rs * sw[d1] + bw[d1];
  x32[row * DD + d0] = o0; x32[row * DD + d1] = o1;
  x16[row * DD + d0] = (f16)o0; x16[row * DD + d1] = (f16)o1;
}

// ---------------- embedding gather + positional encoding ----------------
__global__ __launch_bounds__(256) void embed_kernel(const int* __restrict__ src,
                                                    const float* __restrict__ emb,
                                                    const float* __restrict__ pe,
                                                    float* __restrict__ x32,
                                                    f16* __restrict__ x16) {
  size_t idx = (size_t)blockIdx.x * 256 + threadIdx.x;
  if (idx >= (size_t)BS * DD) return;
  size_t bs = idx / DD;
  int d = (int)(idx % DD);
  int s = (int)(bs % SS);
  float v = emb[(size_t)src[bs] * DD + d] * 22.62741699796952f + pe[(size_t)s * DD + d];
  x32[idx] = v;
  x16[idx] = (f16)v;
}

// ---------------- transpose Vs [B,S,D] -> VsT [B,D,S] (f32) ----------------
__global__ __launch_bounds__(256) void transpose_kernel(const float* __restrict__ in,
                                                        float* __restrict__ out) {
  size_t idx = (size_t)blockIdx.x * 256 + threadIdx.x;
  if (idx >= (size_t)BB * DD * SS) return;
  size_t b = idx / ((size_t)DD * SS);
  size_t r = idx % ((size_t)DD * SS);
  size_t d = r / SS;
  size_t s = r % SS;
  out[idx] = in[b * (size_t)SS * DD + s * DD + d];
}

// ---------------- host orchestration ----------------
extern "C" void kernel_launch(void* const* d_in, const int* in_sizes, int n_in,
                              void* d_out, int out_size, void* d_ws, size_t ws_size,
                              hipStream_t stream) {
  const int*   src    = (const int*)d_in[0];
  const float* emb    = (const float*)d_in[1];
  const float* pe     = (const float*)d_in[2];
  const float* in_w   = (const float*)d_in[3];
  const float* in_b   = (const float*)d_in[4];
  const float* out_w  = (const float*)d_in[5];
  const float* out_b  = (const float*)d_in[6];
  const float* qp_w   = (const float*)d_in[7];
  const float* qp_b   = (const float*)d_in[8];
  const float* kp_w   = (const float*)d_in[9];
  const float* kp_b   = (const float*)d_in[10];
  const float* vp_w   = (const float*)d_in[11];
  const float* vp_b   = (const float*)d_in[12];
  const float* lam    = (const float*)d_in[13];
  const float* ff1_w  = (const float*)d_in[14];
  const float* ff1_b  = (const float*)d_in[15];
  const float* ff2_w  = (const float*)d_in[16];
  const float* ff2_b  = (const float*)d_in[17];
  const float* ln1_s  = (const float*)d_in[18];
  const float* ln1_b  = (const float*)d_in[19];
  const float* ln2_s  = (const float*)d_in[20];
  const float* ln2_b  = (const float*)d_in[21];
  const float* fin_w  = (const float*)d_in[22];
  const float* fin_b  = (const float*)d_in[23];

  char* w = (char*)d_ws;
  size_t off = 0;
  auto alloc = [&](size_t bytes) -> char* {
    char* p = w + off;
    off += (bytes + 255) & ~(size_t)255;
    return p;
  };

  float* x32      = (float*)alloc((size_t)BS * DD * 4);
  f16*   x16      = (f16*)  alloc((size_t)BS * DD * 2);
  f16*   qkv16    = (f16*)  alloc((size_t)BS * 3 * DD * 2);
  f16*   attn16   = (f16*)  alloc((size_t)BS * DD * 2);
  float* dense32  = (float*)alloc((size_t)BS * DD * 4);
  f16*   Qs16     = (f16*)  alloc((size_t)BS * RR * 2);
  float* Ks32     = (float*)alloc((size_t)BS * RR * 4);
  float* Vs32     = (float*)alloc((size_t)BS * DD * 4);
  float* VsT32    = (float*)alloc((size_t)BS * DD * 4);
  float* sparse32 = (float*)alloc((size_t)BS * DD * 4);
  size_t sz_qk = (size_t)BB * SS * SS * 4;
  size_t sz_h  = (size_t)BS * FFF * 2;
  size_t uniA = sz_qk + (size_t)BB * SS * SS * 2;
  size_t uniB = sz_h + (size_t)BS * DD * 4;
  char* uni = alloc(uniA > uniB ? uniA : uniB);
  float* qk32 = (float*)uni;
  f16*   sp16 = (f16*)(uni + sz_qk);
  f16*   h16  = (f16*)uni;
  float* ff32 = (float*)(uni + sz_h);

  auto gemm_f16 = [&](const f16* A, const float* W, const float* bias, f16* O,
                      int M, int N, int K, float scale, bool relu) {
    dim3 g(M / 32, (N + 511) / 512);
    if (relu) gemm_kernel<true, true><<<g, 256, 0, stream>>>(A, W, bias, O, M, N, K, scale);
    else      gemm_kernel<true, false><<<g, 256, 0, stream>>>(A, W, bias, O, M, N, K, scale);
  };
  auto gemm_f32 = [&](const f16* A, const float* W, const float* bias, float* O,
                      int M, int N, int K, float scale) {
    dim3 g(M / 32, (N + 511) / 512);
    gemm_kernel<false, false><<<g, 256, 0, stream>>>(A, W, bias, O, M, N, K, scale);
  };

  {
    size_t tot = (size_t)BS * DD;
    embed_kernel<<<(tot + 255) / 256, 256, 0, stream>>>(src, emb, pe, x32, x16);
  }

  for (int l = 0; l < LL; ++l) {
    const float* inW  = in_w  + (size_t)l * 3 * DD * DD;
    const float* inB  = in_b  + (size_t)l * 3 * DD;
    const float* oW   = out_w + (size_t)l * DD * DD;
    const float* oB   = out_b + (size_t)l * DD;
    const float* qW   = qp_w  + (size_t)l * RR * DD;
    const float* qB   = qp_b  + (size_t)l * RR;
    const float* kW   = kp_w  + (size_t)l * RR * DD;
    const float* kB   = kp_b  + (size_t)l * RR;
    const float* vW   = vp_w  + (size_t)l * DD * DD;
    const float* vB   = vp_b  + (size_t)l * DD;
    const float* f1W  = ff1_w + (size_t)l * FFF * DD;
    const float* f1B  = ff1_b + (size_t)l * FFF;
    const float* f2W  = ff2_w + (size_t)l * DD * FFF;
    const float* f2B  = ff2_b + (size_t)l * DD;

    // dense MHA
    gemm_f16(x16, inW, inB, qkv16, BS, 3 * DD, DD, 1.0f, false);
    attn_kernel<<<dim3(SS / 16, HH, BB), 32, 0, stream>>>(qkv16, attn16);
    gemm_f32(attn16, oW, oB, dense32, BS, DD, DD, 1.0f);

    // low-rank sparse path
    gemm_f16(x16, qW, qB, Qs16, BS, RR, DD, 1.0f, false);
    gemm_f32(x16, kW, kB, Ks32, BS, RR, DD, 1.0f);
    gemm_f32(x16, vW, vB, Vs32, BS, DD, DD, 1.0f);
    {
      size_t tot = (size_t)BB * DD * SS;
      transpose_kernel<<<(tot + 255) / 256, 256, 0, stream>>>(Vs32, VsT32);
    }
    for (int b = 0; b < BB; ++b)
      gemm_f32(Qs16 + (size_t)b * SS * RR, Ks32 + (size_t)b * SS * RR, nullptr,
               qk32 + (size_t)b * SS * SS, SS, SS, RR, 0.125f);
    topk_kernel<<<BS, 256, 0, stream>>>(qk32, sp16);
    for (int b = 0; b < BB; ++b)
      gemm_f32(sp16 + (size_t)b * SS * SS, VsT32 + (size_t)b * DD * SS, nullptr,
               sparse32 + (size_t)b * SS * DD, SS, DD, SS, 1.0f);

    // gated fusion + LN1
    lnfuse_kernel<<<BS, 256, 0, stream>>>(x32, x16, dense32, sparse32, lam, l, 1,
                                          ln1_s + (size_t)l * DD, ln1_b + (size_t)l * DD);
    // FFN + LN2
    gemm_f16(x16, f1W, f1B, h16, BS, FFF, DD, 1.0f, true);
    gemm_f32(h16, f2W, f2B, ff32, BS, DD, FFF, 1.0f);
    lnfuse_kernel<<<BS, 256, 0, stream>>>(x32, x16, ff32, nullptr, nullptr, 0, 0,
                                          ln2_s + (size_t)l * DD, ln2_b + (size_t)l * DD);
  }

  // final vocab projection -> d_out (f32)
  gemm_f32(x16, fin_w, fin_b, (float*)d_out, BS, VV, DD, 1.0f);
  (void)in_sizes; (void)n_in; (void)out_size; (void)ws_size;
}
